// Attention_63299228009317
// MI455X (gfx1250) — matrix operand order repaired and checked by reading, not compiled, not run on hardware
//
#include <hip/hip_runtime.h>
#include <hip/hip_bf16.h>

// ---------------------------------------------------------------------------
// Window-attention block for MI455X (gfx1250), bf16 WMMA pipeline:
//   k0: f32 -> bf16 conversion (x, qkv_w, proj_w)
//   k1: QKV GEMM (bf16 WMMA, f32 acc, reg double-buffered) -> q/k/v^T bf16
//   k2: flash attention per (b,h): K/V tiles async-staged to LDS (double
//       buffered), QK^T WMMA + bias + online softmax, P via LDS relayout,
//       PV WMMA -> bf16 [B,N,C]
//   k3: output projection GEMM (bf16 WMMA, reg double-buffered) + bias -> f32
// ---------------------------------------------------------------------------

typedef __bf16 bf16_t;
typedef __attribute__((ext_vector_type(16))) __bf16 v16bf;
typedef __attribute__((ext_vector_type(8)))  float  v8f;
typedef __attribute__((ext_vector_type(4)))  int    vi4;

union ABFrag { v16bf v; unsigned int u[8]; };

#define AS1 __attribute__((address_space(1)))
#define AS3 __attribute__((address_space(3)))

#if defined(__gfx1250__) && __has_builtin(__builtin_amdgcn_global_load_async_to_lds_b128) && __has_builtin(__builtin_amdgcn_s_wait_asynccnt)
#define USE_ASYNC_LDS 1
#else
#define USE_ASYNC_LDS 0
#endif

__device__ __forceinline__ unsigned short f2bf(float f) {
    unsigned int u = __builtin_bit_cast(unsigned int, f);
    u += 0x7FFFu + ((u >> 16) & 1u);          // round-to-nearest-even
    return (unsigned short)(u >> 16);
}

// A-fragment (16x32 bf16) K-offset of dword pair p for this half-wave.
__device__ __forceinline__ int a_k0(int p, bool hi) {
    return (p << 1) + (hi ? 16 : 0);
}

// copy 16 bytes global -> LDS (async on gfx1250 when available)
__device__ __forceinline__ void cp16_g2lds(const unsigned short* src,
                                           unsigned short* dst) {
#if USE_ASYNC_LDS
    __builtin_amdgcn_global_load_async_to_lds_b128(
        (AS1 vi4*)src, (AS3 vi4*)dst, 0, 0);
#else
    *(vi4*)dst = *(const vi4*)src;
#endif
}

__device__ __forceinline__ void wait_async_lds() {
#if USE_ASYNC_LDS
    __builtin_amdgcn_s_wait_asynccnt(0);
#endif
}

__global__ void cvt_f32_bf16(const float* __restrict__ src,
                             unsigned short* __restrict__ dst, int n) {
    int i = blockIdx.x * blockDim.x + threadIdx.x;
    if (i < n) dst[i] = f2bf(src[i]);
}

// ---------------------------------------------------------------------------
// k1: [32768 x 768] @ [2304 x 768]^T, per-wave 32x64 tile, double-buffered.
// grid = (128, 36), block = 256 (8 waves)
// ---------------------------------------------------------------------------
__global__ __launch_bounds__(256)
void qkv_gemm(const unsigned short* __restrict__ xb,
              const unsigned short* __restrict__ wq,
              unsigned short* __restrict__ qb,
              unsigned short* __restrict__ kb,
              unsigned short* __restrict__ vt) {
    const int lane = threadIdx.x & 31;
    const int wv   = threadIdx.x >> 5;
    const bool hi  = lane >= 16;
    const int l15  = lane & 15;
    const int hi8  = hi ? 8 : 0;
    const int M0   = blockIdx.x * 256 + wv * 32;
    const int N0   = blockIdx.y * 64;

    v8f acc[2][4];
#pragma unroll
    for (int a = 0; a < 2; ++a)
#pragma unroll
        for (int bidx = 0; bidx < 4; ++bidx)
#pragma unroll
            for (int i = 0; i < 8; ++i) acc[a][bidx][i] = 0.0f;

    const unsigned short* arow0 = xb + (M0 + l15) * 768;
    const unsigned short* arow1 = xb + (M0 + 16 + l15) * 768;

    ABFrag A0[2], A1[2], Bf[2][4];
    auto load_tile = [&](int buf, int kk) {
#pragma unroll
        for (int p = 0; p < 8; ++p) {
            const int k0 = a_k0(p, hi);
            A0[buf].u[p] = *(const unsigned int*)(arow0 + kk + k0);
            A1[buf].u[p] = *(const unsigned int*)(arow1 + kk + k0);
        }
        const int bofs = kk + (hi ? 16 : 0);
#pragma unroll
        for (int nf = 0; nf < 4; ++nf) {
            const unsigned short* brow = wq + (N0 + nf * 16 + l15) * 768 + bofs;
#pragma unroll
            for (int p = 0; p < 8; ++p)
                Bf[buf][nf].u[p] = *(const unsigned int*)(brow + p * 2);
        }
    };

    load_tile(0, 0);
#pragma unroll
    for (int it = 0; it < 24; ++it) {
        const int cur = it & 1;
        if (it + 1 < 24) load_tile(cur ^ 1, (it + 1) * 32);
#pragma unroll
        for (int nf = 0; nf < 4; ++nf) {
            acc[0][nf] = __builtin_amdgcn_wmma_f32_16x16x32_bf16(
                false, A0[cur].v, false, Bf[cur][nf].v, (short)0, acc[0][nf], false, false);
            acc[1][nf] = __builtin_amdgcn_wmma_f32_16x16x32_bf16(
                false, A1[cur].v, false, Bf[cur][nf].v, (short)0, acc[1][nf], false, false);
        }
    }

    // Scatter into q (scaled, [bh][n][64]), k ([bh][n][64]), v^T ([bh][d][512])
#pragma unroll
    for (int nf = 0; nf < 4; ++nf) {
        const int n     = N0 + nf * 16 + l15;
        const int which = n / 768;           // 0=q 1=k 2=v
        const int c     = n - which * 768;
        const int h     = c >> 6;
        const int d     = c & 63;
#pragma unroll
        for (int mf = 0; mf < 2; ++mf) {
#pragma unroll
            for (int i = 0; i < 8; ++i) {
                const int m  = M0 + mf * 16 + i + hi8;
                const int bb = m >> 9;
                const int nr = m & 511;
                const int bh = bb * 12 + h;
                const float val = acc[mf][nf][i];
                if (which == 0)
                    qb[(bh * 512 + nr) * 64 + d] = f2bf(val * 0.125f); // *hd^-0.5
                else if (which == 1)
                    kb[(bh * 512 + nr) * 64 + d] = f2bf(val);
                else
                    vt[(bh * 64 + d) * 512 + nr] = f2bf(val);
            }
        }
    }
}

// ---------------------------------------------------------------------------
// k2: flash attention. grid = 768*4 blocks, block = 256 (8 waves).
// Each wave: 16 query rows; 32-key tiles; K/V tiles double-buffered in LDS
// via async global->LDS copies shared by all 8 waves.
// ---------------------------------------------------------------------------
__global__ __launch_bounds__(256)
void attn_kernel(const unsigned short* __restrict__ qb,
                 const unsigned short* __restrict__ kb,
                 const unsigned short* __restrict__ vt,
                 const float* __restrict__ bias_table,
                 unsigned short* __restrict__ ob) {
    __shared__ float s_bias[1023];
    __shared__ __align__(16) unsigned short s_p[8 * 16 * 32];
    __shared__ __align__(16) unsigned short s_k[2][32 * 64]; // [key][d]
    __shared__ __align__(16) unsigned short s_v[2][64 * 32]; // [d][key]

    const int bh = blockIdx.x >> 2;
    const int qt = blockIdx.x & 3;
    const int h  = bh % 12;
    const int b  = bh / 12;
    const int tid  = threadIdx.x;
    const int lane = tid & 31;
    const int wv   = tid >> 5;
    const bool hi  = lane >= 16;
    const int l15  = lane & 15;
    const int hi8  = hi ? 8 : 0;
    const int r0   = qt * 128 + wv * 16;      // global query-row base of this wave

    // stage bias column h (1023 floats) into LDS
    for (int i = tid; i < 1023; i += 256)
        s_bias[i] = bias_table[i * 12 + h];

    // stage K tile (32x64, contiguous 4KB) and V tile (64 rows x 32 keys)
    const int vrow_d = tid >> 2;            // 0..63
    const int vseg   = (tid & 3) * 8;       // 0,8,16,24 elements (16B)
    auto stage_tiles = [&](int buf, int key0) {
        cp16_g2lds(kb + (bh * 512 + key0) * 64 + tid * 8, &s_k[buf][tid * 8]);
        cp16_g2lds(vt + (bh * 64 + vrow_d) * 512 + key0 + vseg,
                   &s_v[buf][vrow_d * 32 + vseg]);
    };

    stage_tiles(0, 0);

    // Q fragments (16x64 split into two 16x32 A-frags), pre-scaled
    ABFrag qf[2];
    {
        const unsigned short* qrow = qb + (bh * 512 + r0 + l15) * 64;
#pragma unroll
        for (int c = 0; c < 2; ++c)
#pragma unroll
            for (int p = 0; p < 8; ++p)
                qf[c].u[p] = *(const unsigned int*)(qrow + c * 32 + a_k0(p, hi));
    }

    v8f acc[4];
#pragma unroll
    for (int ch = 0; ch < 4; ++ch)
#pragma unroll
        for (int i = 0; i < 8; ++i) acc[ch][i] = 0.0f;
    float m_i[8], l_i[8];
#pragma unroll
    for (int i = 0; i < 8; ++i) { m_i[i] = -__builtin_inff(); l_i[i] = 0.0f; }

    unsigned short* pbuf = s_p + wv * (16 * 32);

    wait_async_lds();
    __syncthreads();          // bias + first K/V tile visible

    for (int jj = 0; jj < 16; ++jj) {
        const int key0 = jj * 32;
        const int cur  = jj & 1;

        // kick off next tile's async copies into the other LDS buffer
        if (jj + 1 < 16) stage_tiles(cur ^ 1, key0 + 32);

        // S = Q @ K^T for two 16-key sub-tiles, + relative-position bias
        v8f s[2];
#pragma unroll
        for (int t = 0; t < 2; ++t) {
            v8f st;
#pragma unroll
            for (int i = 0; i < 8; ++i) st[i] = 0.0f;
            const int kcol = key0 + t * 16 + l15;
            const unsigned short* krow =
                &s_k[cur][(t * 16 + l15) * 64] + (hi ? 16 : 0);
#pragma unroll
            for (int c = 0; c < 2; ++c) {
                ABFrag kf;
#pragma unroll
                for (int p = 0; p < 8; ++p)
                    kf.u[p] = *(const unsigned int*)(krow + c * 32 + p * 2);
                st = __builtin_amdgcn_wmma_f32_16x16x32_bf16(
                    false, qf[c].v, false, kf.v, (short)0, st, false, false);
            }
#pragma unroll
            for (int i = 0; i < 8; ++i)
                st[i] += s_bias[(r0 + i + hi8) - kcol + 511];
            s[t] = st;
        }

        // online softmax (row reduction across the 16 lanes holding that row)
        float p0[8], p1[8];
#pragma unroll
        for (int i = 0; i < 8; ++i) {
            float tm = fmaxf(s[0][i], s[1][i]);
#pragma unroll
            for (int off = 1; off < 16; off <<= 1)
                tm = fmaxf(tm, __shfl_xor(tm, off, 32));
            const float mn    = fmaxf(m_i[i], tm);
            const float alpha = __expf(m_i[i] - mn);
            const float e0 = __expf(s[0][i] - mn);
            const float e1 = __expf(s[1][i] - mn);
            float rs = e0 + e1;
#pragma unroll
            for (int off = 1; off < 16; off <<= 1)
                rs += __shfl_xor(rs, off, 32);
            l_i[i] = l_i[i] * alpha + rs;
            m_i[i] = mn;
#pragma unroll
            for (int ch = 0; ch < 4; ++ch) acc[ch][i] *= alpha;
            p0[i] = e0; p1[i] = e1;
        }

        // P (C-layout) -> LDS row-major 16x32 -> reload in A-layout
#pragma unroll
        for (int i = 0; i < 8; ++i) {
            const int row = i + hi8;
            pbuf[row * 32 + l15]      = f2bf(p0[i]);
            pbuf[row * 32 + 16 + l15] = f2bf(p1[i]);
        }
        __syncthreads();
        ABFrag pf;
#pragma unroll
        for (int p = 0; p < 8; ++p)
            pf.u[p] = *(const unsigned int*)(pbuf + l15 * 32 + a_k0(p, hi));

        // O += P @ V from LDS V tile
#pragma unroll
        for (int ch = 0; ch < 4; ++ch) {
            ABFrag vf;
            const unsigned short* vrow =
                &s_v[cur][(ch * 16 + l15) * 32] + (hi ? 16 : 0);
#pragma unroll
            for (int p = 0; p < 8; ++p)
                vf.u[p] = *(const unsigned int*)(vrow + p * 2);
            acc[ch] = __builtin_amdgcn_wmma_f32_16x16x32_bf16(
                false, pf.v, false, vf.v, (short)0, acc[ch], false, false);
        }

        // next tile's async copies must be complete & visible to all waves,
        // and P buffer free for next iteration
        wait_async_lds();
        __syncthreads();
    }

    // normalize and store bf16 attention output as [B,N,C]
#pragma unroll
    for (int i = 0; i < 8; ++i) l_i[i] = 1.0f / l_i[i];
#pragma unroll
    for (int ch = 0; ch < 4; ++ch)
#pragma unroll
        for (int i = 0; i < 8; ++i) {
            const int row = r0 + i + hi8;
            ob[(b * 512 + row) * 768 + h * 64 + ch * 16 + l15] =
                f2bf(acc[ch][i] * l_i[i]);
        }
}

// ---------------------------------------------------------------------------
// k3: out = ob @ proj_w^T + proj_b, f32 result. grid = (128, 12), block = 256.
// ---------------------------------------------------------------------------
__global__ __launch_bounds__(256)
void proj_gemm(const unsigned short* __restrict__ ob,
               const unsigned short* __restrict__ wp,
               const float* __restrict__ pbias,
               float* __restrict__ out) {
    const int lane = threadIdx.x & 31;
    const int wv   = threadIdx.x >> 5;
    const bool hi  = lane >= 16;
    const int l15  = lane & 15;
    const int hi8  = hi ? 8 : 0;
    const int M0   = blockIdx.x * 256 + wv * 32;
    const int N0   = blockIdx.y * 64;

    v8f acc[2][4];
#pragma unroll
    for (int a = 0; a < 2; ++a)
#pragma unroll
        for (int bidx = 0; bidx < 4; ++bidx)
#pragma unroll
            for (int i = 0; i < 8; ++i) acc[a][bidx][i] = 0.0f;

    const unsigned short* arow0 = ob + (M0 + l15) * 768;
    const unsigned short* arow1 = ob + (M0 + 16 + l15) * 768;

    ABFrag A0[2], A1[2], Bf[2][4];
    auto load_tile = [&](int buf, int kk) {
#pragma unroll
        for (int p = 0; p < 8; ++p) {
            const int k0 = a_k0(p, hi);
            A0[buf].u[p] = *(const unsigned int*)(arow0 + kk + k0);
            A1[buf].u[p] = *(const unsigned int*)(arow1 + kk + k0);
        }
        const int bofs = kk + (hi ? 16 : 0);
#pragma unroll
        for (int nf = 0; nf < 4; ++nf) {
            const unsigned short* brow = wp + (N0 + nf * 16 + l15) * 768 + bofs;
#pragma unroll
            for (int p = 0; p < 8; ++p)
                Bf[buf][nf].u[p] = *(const unsigned int*)(brow + p * 2);
        }
    };

    load_tile(0, 0);
#pragma unroll
    for (int it = 0; it < 24; ++it) {
        const int cur = it & 1;
        if (it + 1 < 24) load_tile(cur ^ 1, (it + 1) * 32);
#pragma unroll
        for (int nf = 0; nf < 4; ++nf) {
            acc[0][nf] = __builtin_amdgcn_wmma_f32_16x16x32_bf16(
                false, A0[cur].v, false, Bf[cur][nf].v, (short)0, acc[0][nf], false, false);
            acc[1][nf] = __builtin_amdgcn_wmma_f32_16x16x32_bf16(
                false, A1[cur].v, false, Bf[cur][nf].v, (short)0, acc[1][nf], false, false);
        }
    }

#pragma unroll
    for (int nf = 0; nf < 4; ++nf) {
        const int n = N0 + nf * 16 + l15;
        const float bn = pbias[n];
#pragma unroll
        for (int mf = 0; mf < 2; ++mf)
#pragma unroll
            for (int i = 0; i < 8; ++i) {
                const int m = M0 + mf * 16 + i + hi8;
                out[m * 768 + n] = acc[mf][nf][i] + bn;
            }
    }
}

// ---------------------------------------------------------------------------
extern "C" void kernel_launch(void* const* d_in, const int* in_sizes, int n_in,
                              void* d_out, int out_size, void* d_ws, size_t ws_size,
                              hipStream_t stream) {
    const float* x          = (const float*)d_in[0]; // [64,512,768]
    const float* qkv_w      = (const float*)d_in[1]; // [2304,768]
    const float* proj_w     = (const float*)d_in[2]; // [768,768]
    const float* proj_b     = (const float*)d_in[3]; // [768]
    const float* bias_table = (const float*)d_in[4]; // [1023,12]
    float* out = (float*)d_out;

    const size_t N_X  = 32768ull * 768;   // 25,165,824
    const size_t N_WQ = 2304ull * 768;
    const size_t N_WP = 768ull * 768;
    const size_t N_HD = 768ull * 512 * 64; // per q/k/vt buffer

    size_t off = 0;
    auto take = [&](size_t elems) {
        unsigned short* p = (unsigned short*)((char*)d_ws + off);
        off = (off + elems * 2 + 255) & ~(size_t)255;
        return p;
    };
    unsigned short* xb  = take(N_X);
    unsigned short* wqb = take(N_WQ);
    unsigned short* wpb = take(N_WP);
    unsigned short* qb  = take(N_HD);
    unsigned short* kb  = take(N_HD);
    unsigned short* vt  = take(N_HD);
    unsigned short* ob  = take(N_X);
    (void)ws_size; (void)in_sizes; (void)n_in; (void)out_size;

    cvt_f32_bf16<<<(int)((N_X  + 255) / 256), 256, 0, stream>>>(x,      xb,  (int)N_X);
    cvt_f32_bf16<<<(int)((N_WQ + 255) / 256), 256, 0, stream>>>(qkv_w,  wqb, (int)N_WQ);
    cvt_f32_bf16<<<(int)((N_WP + 255) / 256), 256, 0, stream>>>(proj_w, wpb, (int)N_WP);

    qkv_gemm<<<dim3(128, 36), 256, 0, stream>>>(xb, wqb, qb, kb, vt);
    attn_kernel<<<dim3(768 * 4), 256, 0, stream>>>(qb, kb, vt, bias_table, ob);
    proj_gemm<<<dim3(128, 12), 256, 0, stream>>>(ob, wpb, proj_b, out);
}
